// FresnelWaveAttention_1906965479947
// MI455X (gfx1250) — matrix operand order repaired and checked by reading, not compiled, not run on hardware
//
#include <hip/hip_runtime.h>
#include <math.h>

typedef __attribute__((ext_vector_type(16))) _Float16 v16h;
typedef __attribute__((ext_vector_type(8)))  _Float16 v8h;
typedef __attribute__((ext_vector_type(8)))  float    v8f;

#define NTOK 2304   // 48*48
#define CDIM 512
#define D3   1536
#define NH   8
#define DH   64
#define HW   48

// ---------------------------------------------------------------------------
// WMMA fragment helpers (CDNA5 wave32, v_wmma_f32_16x16x32_f16)
//
// A (16x32 f16): lane&15 = M row; element j holds K = 16*(j>>3) + 8*(lane>>4) + (j&7)
//   -> per lane: two contiguous 8-half (16B) runs at K-offsets {8*hi} and {16+8*hi}
// B (32x16 f16): lane&15 = N col; element j holds K = 16*(j>>3) + 8*(lane>>4) + (j&7)
//   -> per lane: two contiguous 8-half (16B) runs from B^T row-major storage
// C/D (16x16 f32): lane&15 = N col; element r holds M = r + 8*(lane>>4)
// ---------------------------------------------------------------------------

static __device__ __forceinline__ v16h load_a_frag(const _Float16* p, int ld) {
  const int lane = threadIdx.x & 31;
  const _Float16* r = p + (lane & 15) * ld + ((lane >> 4) << 3);
  v8h lo = *(const v8h*)(r);
  v8h hi = *(const v8h*)(r + 16);
  v16h a;
#pragma unroll
  for (int j = 0; j < 8; ++j) { a[j] = lo[j]; a[8 + j] = hi[j]; }
  return a;
}

// pT: pointer to element (k=0,n=0) of a 32x16 B tile stored as B^T row-major
// (i.e. k contiguous per n-row), leading dimension ldT halfs.
static __device__ __forceinline__ v16h load_b_frag_t(const _Float16* pT, int ldT) {
  const int lane = threadIdx.x & 31;
  const _Float16* r = pT + (lane & 15) * ldT + ((lane >> 4) << 3);
  v8h lo = *(const v8h*)(r);
  v8h hi = *(const v8h*)(r + 16);
  v16h b;
#pragma unroll
  for (int j = 0; j < 8; ++j) { b[j] = lo[j]; b[8 + j] = hi[j]; }
  return b;
}

static __device__ __forceinline__ v8f wmma32(v16h a, v16h b, v8f c) {
  return __builtin_amdgcn_wmma_f32_16x16x32_f16(false, a, false, b, (short)0, c,
                                                false, false);
}

// ---------------------------------------------------------------------------
// Kernel 0: f32 -> f16 weight conversion
// ---------------------------------------------------------------------------
__global__ __launch_bounds__(256) void cvt_f16_kernel(const float* __restrict__ src,
                                                      _Float16* __restrict__ dst,
                                                      int n) {
  int i = blockIdx.x * 256 + threadIdx.x;
  if (i < n) dst[i] = (_Float16)src[i];
}

// ---------------------------------------------------------------------------
// Kernel 1: Fresnel interference bias, bias[i*N+j] = 0.1*cos(2*pi*d_ij/denom)
// Computed once; 21 MB lives in the 192 MB L2 and is reused by all 16 (b,h).
// ---------------------------------------------------------------------------
__global__ __launch_bounds__(256) void fresnel_bias_kernel(
    const float* __restrict__ wavelength, float* __restrict__ bias) {
  int idx = blockIdx.x * 256 + threadIdx.x;
  if (idx >= NTOK * NTOK) return;
  int i = idx / NTOK, j = idx % NTOK;
  float dy = (float)(i / HW - j / HW);
  float dx = (float)(i % HW - j % HW);
  float dist = sqrtf(dy * dy + dx * dx + 1e-8f);
  float denom = fabsf(wavelength[0]) * (float)HW + 1e-6f;
  float phase = 6.28318530717958647692f * dist / denom;
  bias[idx] = 0.1f * cosf(phase);
}

// ---------------------------------------------------------------------------
// Kernel 2: QKV projection. qkv[b,n,d] = sum_c x[b,c,n] * w_qkv[d,c].
// 64x64 output tile per block (4 waves, each 16 rows x 64 cols).
// X tile staged through LDS (transposed f32->f16); W fragments loaded
// directly from global (rows are K-contiguous). Since the 64-wide tile is
// aligned to both the 512-wide q/k/v sections and 64-wide heads, sec/head
// are block-uniform -> branch-free epilogue.
// Writes Q (pre-scaled by dh^-0.5) and K as [B,NH,N,DH] f16; V transposed
// as [B,NH,DH,N] f16 so the flash kernel's P*V B-fragments are contiguous.
// ---------------------------------------------------------------------------
__global__ __launch_bounds__(128) void qkv_gemm_kernel(
    const float* __restrict__ x, const _Float16* __restrict__ wq16,
    _Float16* __restrict__ qh, _Float16* __restrict__ kh,
    _Float16* __restrict__ vt) {
  __shared__ _Float16 lds_a[64][40];  // 40-half (80B) pitch, 16B aligned rows
  const int b = blockIdx.z;
  const int n0 = blockIdx.x * 64;
  const int j0 = blockIdx.y * 64;
  const int t = threadIdx.x;
  const int wave = t >> 5, lane = t & 31;
  const int colb = lane & 15, rhi = (lane >> 4) << 3;
  const float* xb = x + (size_t)b * CDIM * NTOK;

  v8f acc[4] = {};
  for (int k0 = 0; k0 < CDIM; k0 += 32) {
#pragma unroll
    for (int i = 0; i < 16; ++i) {
      int idx = t + 128 * i;      // 0..2047
      int cl = idx >> 6;          // 0..31 (K)
      int nl = idx & 63;          // 0..63 (M) -> coalesced global reads
      lds_a[nl][cl] = (_Float16)xb[(size_t)(k0 + cl) * NTOK + n0 + nl];
    }
    __syncthreads();
    v16h a = load_a_frag(&lds_a[wave * 16][0], 40);
#pragma unroll
    for (int s = 0; s < 4; ++s) {
      v16h bb = load_b_frag_t(wq16 + (size_t)(j0 + s * 16) * CDIM + k0, CDIM);
      acc[s] = wmma32(a, bb, acc[s]);
    }
    __syncthreads();
  }

  // Block-uniform section/head selection (j0 is 64-aligned; sections are
  // 512-aligned, heads 64-aligned).
  const int sec = j0 >> 9;                 // 0=q, 1=k, 2=v (uniform)
  const int head = (j0 & 511) >> 6;        // uniform
  const size_t hb = (size_t)b * NH + head;
  const int nrow = n0 + wave * 16 + rhi;

  if (sec == 0) {
#pragma unroll
    for (int s = 0; s < 4; ++s) {
      int dd = s * 16 + colb;
#pragma unroll
      for (int r = 0; r < 8; ++r)
        qh[(hb * NTOK + nrow + r) * DH + dd] =
            (_Float16)(acc[s][r] * 0.125f);  // * dh^-0.5
    }
  } else if (sec == 1) {
#pragma unroll
    for (int s = 0; s < 4; ++s) {
      int dd = s * 16 + colb;
#pragma unroll
      for (int r = 0; r < 8; ++r)
        kh[(hb * NTOK + nrow + r) * DH + dd] = (_Float16)acc[s][r];
    }
  } else {
#pragma unroll
    for (int s = 0; s < 4; ++s) {
      int dd = s * 16 + colb;
#pragma unroll
      for (int r = 0; r < 8; ++r)
        vt[(hb * DH + dd) * NTOK + nrow + r] = (_Float16)acc[s][r];
    }
  }
}

// ---------------------------------------------------------------------------
// Kernel 3: flash attention per (b, h). Block = 64 query rows (4 waves x 16).
// Per 64-key block: S = Q*K^T (+bias) via WMMA, online softmax in f32
// registers (half-wave shfl reductions), P staged through a per-wave LDS
// strip to re-shape C-layout -> A-layout, then O += P*V via WMMA on the
// transposed V. Next key-block's K/V lines are prefetched (global_prefetch)
// while the current block computes. Never materializes the N x N matrix.
// ---------------------------------------------------------------------------
__global__ __launch_bounds__(128) void flash_attn_kernel(
    const _Float16* __restrict__ qh, const _Float16* __restrict__ kh,
    const _Float16* __restrict__ vt, const float* __restrict__ bias,
    _Float16* __restrict__ att) {
  __shared__ _Float16 lds_p[4][16][72];  // per-wave P strip, 144B pitch
  const int b = blockIdx.z, h = blockIdx.y;
  const int q0 = blockIdx.x * 64;
  const int wave = threadIdx.x >> 5;
  const int lane = threadIdx.x & 31;
  const int colb = lane & 15;
  const int rhi = (lane >> 4) << 3;
  const size_t bh = (size_t)b * NH + h;
  const _Float16* qptr = qh + (bh * NTOK + q0 + wave * 16) * DH;
  const _Float16* kbh = kh + bh * NTOK * DH;
  const _Float16* vbh = vt + bh * DH * NTOK;
  const int qrow_base = q0 + wave * 16 + rhi;
  const int t64 = threadIdx.x & 63;  // 0..63 row id for prefetch

  v16h aq0 = load_a_frag(qptr, DH);        // Q rows, K = 0..31
  v16h aq1 = load_a_frag(qptr + 32, DH);   // Q rows, K = 32..63

  v8f o[4] = {};
  float mrow[8], lrow[8];
#pragma unroll
  for (int r = 0; r < 8; ++r) { mrow[r] = -1e30f; lrow[r] = 0.f; }

  for (int jb = 0; jb < NTOK; jb += 64) {
    // Prefetch next key block: each K row is 128B (one line); each V row
    // slice is 128B. One byte-prefetch per row pulls the whole line.
    int jn = jb + 64;
    if (jn < NTOK) {
      __builtin_prefetch(kbh + (size_t)(jn + t64) * DH, 0, 1);
      __builtin_prefetch(vbh + (size_t)t64 * NTOK + jn, 0, 1);
    }

    v8f s[4];
#pragma unroll
    for (int s4 = 0; s4 < 4; ++s4) {
      v8f acc = {};
      v16h bk0 = load_b_frag_t(kbh + (size_t)(jb + s4 * 16) * DH, DH);
      acc = wmma32(aq0, bk0, acc);
      v16h bk1 = load_b_frag_t(kbh + (size_t)(jb + s4 * 16) * DH + 32, DH);
      acc = wmma32(aq1, bk1, acc);
#pragma unroll
      for (int r = 0; r < 8; ++r)
        acc[r] += bias[(size_t)(qrow_base + r) * NTOK + jb + s4 * 16 + colb];
      s[s4] = acc;
    }

    // online softmax per row (row lives in 16 lanes of a half-wave)
#pragma unroll
    for (int r = 0; r < 8; ++r) {
      float mx = fmaxf(fmaxf(s[0][r], s[1][r]), fmaxf(s[2][r], s[3][r]));
      mx = fmaxf(mx, __shfl_xor(mx, 1, 16));
      mx = fmaxf(mx, __shfl_xor(mx, 2, 16));
      mx = fmaxf(mx, __shfl_xor(mx, 4, 16));
      mx = fmaxf(mx, __shfl_xor(mx, 8, 16));
      float mnew = fmaxf(mrow[r], mx);
      float corr = __expf(mrow[r] - mnew);
      float rsum = 0.f;
#pragma unroll
      for (int s4 = 0; s4 < 4; ++s4) {
        float p = __expf(s[s4][r] - mnew);
        s[s4][r] = p;
        rsum += p;
      }
      rsum += __shfl_xor(rsum, 1, 16);
      rsum += __shfl_xor(rsum, 2, 16);
      rsum += __shfl_xor(rsum, 4, 16);
      rsum += __shfl_xor(rsum, 8, 16);
      lrow[r] = lrow[r] * corr + rsum;
      mrow[r] = mnew;
#pragma unroll
      for (int s2 = 0; s2 < 4; ++s2) o[s2][r] *= corr;
    }

    // stash P (C-layout -> LDS row-major) then reload as A-fragments
#pragma unroll
    for (int s4 = 0; s4 < 4; ++s4)
#pragma unroll
      for (int r = 0; r < 8; ++r)
        lds_p[wave][rhi + r][s4 * 16 + colb] = (_Float16)s[s4][r];
    __syncthreads();
    const _Float16* pp = &lds_p[wave][0][0];
    v16h ap0 = load_a_frag(pp, 72);
    v16h ap1 = load_a_frag(pp + 32, 72);
#pragma unroll
    for (int s2 = 0; s2 < 4; ++s2) {
      v16h bv0 = load_b_frag_t(vbh + (size_t)(s2 * 16) * NTOK + jb, NTOK);
      o[s2] = wmma32(ap0, bv0, o[s2]);
      v16h bv1 = load_b_frag_t(vbh + (size_t)(s2 * 16) * NTOK + jb + 32, NTOK);
      o[s2] = wmma32(ap1, bv1, o[s2]);
    }
    __syncthreads();
  }

  // normalize and emit attention output as f16 [B, N, 512]
#pragma unroll
  for (int r = 0; r < 8; ++r) {
    float inv = 1.0f / lrow[r];
    int ng = qrow_base + r;
#pragma unroll
    for (int s2 = 0; s2 < 4; ++s2)
      att[((size_t)b * NTOK + ng) * CDIM + h * DH + s2 * 16 + colb] =
          (_Float16)(o[s2][r] * inv);
  }
}

// ---------------------------------------------------------------------------
// Kernel 4: output projection. out[b,d,n] = sum_c att[b,n,c]*w_out[d,c]+b_out[d]
// Both operands f16 with K contiguous -> pure global fragment loads. Epilogue
// transposes through LDS so the [B,C,N] store is n-contiguous (coalesced).
// ---------------------------------------------------------------------------
__global__ __launch_bounds__(128) void out_proj_kernel(
    const _Float16* __restrict__ att, const _Float16* __restrict__ wo16,
    const float* __restrict__ b_out, float* __restrict__ out) {
  __shared__ float lds_o[64][65];
  const int b = blockIdx.z;
  const int n0 = blockIdx.x * 64;
  const int j0 = blockIdx.y * 64;
  const int t = threadIdx.x;
  const int wave = t >> 5, lane = t & 31;
  const int colb = lane & 15, rhi = (lane >> 4) << 3;
  const _Float16* abase = att + ((size_t)b * NTOK + n0 + wave * 16) * CDIM;

  v8f acc[4] = {};
  for (int k0 = 0; k0 < CDIM; k0 += 32) {
    v16h a = load_a_frag(abase + k0, CDIM);
#pragma unroll
    for (int s = 0; s < 4; ++s) {
      v16h bb = load_b_frag_t(wo16 + (size_t)(j0 + s * 16) * CDIM + k0, CDIM);
      acc[s] = wmma32(a, bb, acc[s]);
    }
  }

#pragma unroll
  for (int s = 0; s < 4; ++s) {
    int dl = s * 16 + colb;
    float bb = b_out[j0 + dl];
#pragma unroll
    for (int r = 0; r < 8; ++r)
      lds_o[dl][wave * 16 + rhi + r] = acc[s][r] + bb;
  }
  __syncthreads();
  for (int idx = t; idx < 64 * 64; idx += 128) {
    int dl = idx >> 6, nl = idx & 63;
    out[((size_t)b * CDIM + j0 + dl) * NTOK + n0 + nl] = lds_o[dl][nl];
  }
}

// ---------------------------------------------------------------------------
// Host launcher
// ---------------------------------------------------------------------------
extern "C" void kernel_launch(void* const* d_in, const int* in_sizes, int n_in,
                              void* d_out, int out_size, void* d_ws,
                              size_t ws_size, hipStream_t stream) {
  (void)in_sizes; (void)n_in; (void)out_size; (void)ws_size;
  const float* x = (const float*)d_in[0];          // [2,512,48,48]
  const float* w_qkv = (const float*)d_in[1];      // [1536,512]
  const float* w_out = (const float*)d_in[2];      // [512,512]
  const float* b_out = (const float*)d_in[3];      // [512]
  const float* wavelength = (const float*)d_in[4]; // [1]
  float* out = (float*)d_out;                      // [2,512,48,48]

  // Workspace layout (~42 MB total)
  char* ws = (char*)d_ws;
  _Float16* wq16 = (_Float16*)ws;  ws += (size_t)D3 * CDIM * 2;          // 1.5 MB
  _Float16* wo16 = (_Float16*)ws;  ws += (size_t)CDIM * CDIM * 2;        // 0.5 MB
  _Float16* qh   = (_Float16*)ws;  ws += (size_t)2 * NH * NTOK * DH * 2; // 4.7 MB
  _Float16* kh   = (_Float16*)ws;  ws += (size_t)2 * NH * NTOK * DH * 2; // 4.7 MB
  _Float16* vt   = (_Float16*)ws;  ws += (size_t)2 * NH * NTOK * DH * 2; // 4.7 MB
  _Float16* att  = (_Float16*)ws;  ws += (size_t)2 * NTOK * CDIM * 2;    // 4.7 MB
  float*    bias = (float*)ws;     ws += (size_t)NTOK * NTOK * 4;        // 21.2 MB

  cvt_f16_kernel<<<(D3 * CDIM + 255) / 256, 256, 0, stream>>>(w_qkv, wq16,
                                                              D3 * CDIM);
  cvt_f16_kernel<<<(CDIM * CDIM + 255) / 256, 256, 0, stream>>>(w_out, wo16,
                                                                CDIM * CDIM);
  fresnel_bias_kernel<<<(NTOK * NTOK + 255) / 256, 256, 0, stream>>>(wavelength,
                                                                     bias);
  qkv_gemm_kernel<<<dim3(NTOK / 64, D3 / 64, 2), 128, 0, stream>>>(x, wq16, qh,
                                                                   kh, vt);
  flash_attn_kernel<<<dim3(NTOK / 64, NH, 2), 128, 0, stream>>>(qh, kh, vt,
                                                                bias, att);
  out_proj_kernel<<<dim3(NTOK / 64, CDIM / 64, 2), 128, 0, stream>>>(att, wo16,
                                                                     b_out, out);
}
